// MUS_layer_57174604644510
// MI455X (gfx1250) — compile-verified, matrix operand order checked by reading, not run on hardware
//
#include <hip/hip_runtime.h>

#define CCH 128          // channels
#define NT  8            // N tiles per 128-wide output (128/16)

typedef __attribute__((ext_vector_type(16))) __bf16 bf16x16;
typedef __attribute__((ext_vector_type(8)))  float  f32x8;

__device__ __forceinline__ f32x8 wmma_bf16(bf16x16 a, bf16x16 b, f32x8 c) {
  // (neg_a, A, neg_b, B, c_mod, C, reuse_a, reuse_b)
  return __builtin_amdgcn_wmma_f32_16x16x32_bf16(false, a, false, b, (short)0, c, false, false);
}

// ---------------------------------------------------------------------------
// Pack a (K x N) row-major f32 weight matrix into per-wave WMMA B fragments
// (bf16).  Fragment (kt, nt) = 32 lanes x 16 bf16; lane holds column
// n = nt*16 + (lane&15) and k = kt*32 + ((lane>=16)?8:0) + (i<8 ? i : i+8)
// (mirrors the 16-bit A/B 16x16x32 VGPR layout from CDNA5 ISA).
// ---------------------------------------------------------------------------
__global__ void pack_w(const float* __restrict__ W, __bf16* __restrict__ out,
                       int K, int N) {
  int idx = blockIdx.x * blockDim.x + threadIdx.x;
  if (idx >= K * N) return;
  int i    = idx & 15;
  int lane = (idx >> 4) & 31;
  int frag = idx >> 9;
  int nt = frag % (N / 16);
  int kt = frag / (N / 16);
  int k = kt * 32 + ((lane >= 16) ? 8 : 0) + (i < 8 ? i : i + 8);
  int n = nt * 16 + (lane & 15);
  out[idx] = (__bf16)W[(size_t)k * N + n];
}

// ---------------------------------------------------------------------------
// Fused 2-layer MLP with residual:  io += relu([segs]@W1 + b1)@W2 + b2
// NSEG==3: edge MLP, segs = [io(e) | xb[src] | xb[dst]]  (K1 = 384)
// NSEG==2: node MLP, segs = [io(x) | xb(agg)]            (K1 = 256)
// One wave handles one 16-row tile; 4 waves per block, wave-private LDS
// staging for the relu intermediate (D-layout -> A-layout transpose).
// ---------------------------------------------------------------------------
template <int NSEG>
__global__ __launch_bounds__(128) void fused_mlp(
    float* __restrict__ io, const float* __restrict__ xb,
    const int* __restrict__ sIdx, const int* __restrict__ dIdx,
    const __bf16* __restrict__ W1p, const float* __restrict__ b1,
    const __bf16* __restrict__ W2p, const float* __restrict__ b2,
    int nRows)
{
  constexpr int KT1 = NSEG * 4;            // K1/32
  __shared__ __bf16 lds_t[4][16 * CCH];    // 16 KB, one slice per wave

  const int lane = threadIdx.x & 31;
  const int wv   = threadIdx.x >> 5;
  const int tile = blockIdx.x * 4 + wv;
  const int nTiles = (nRows + 15) >> 4;
  if (tile >= nTiles) return;              // wave-uniform exit

  const int row   = lane & 15;             // A-matrix row owned by this lane
  const int khalf = (lane >= 16) ? 8 : 0;  // per-half-wave k offset
  const int mbase = khalf;                 // rows owned in C/D fragments
  const int ncol  = lane & 15;             // C/D column owned by this lane
  int r0 = tile * 16 + row;
  int rc = r0 < nRows ? r0 : nRows - 1;    // clamp gather for ragged tail

  const float* segp[3];
  segp[0] = io + (size_t)rc * CCH;
  if (NSEG == 3) {
    segp[1] = xb + (size_t)sIdx[rc] * CCH;
    segp[2] = xb + (size_t)dIdx[rc] * CCH;
  } else {
    segp[1] = xb + (size_t)rc * CCH;
    segp[2] = segp[1];
  }

  // -------- GEMM1: (16 x K1) @ (K1 x 128), f32 accum --------
  f32x8 acc[NT];
#pragma unroll
  for (int nt = 0; nt < NT; ++nt) acc[nt] = (f32x8){0,0,0,0,0,0,0,0};

#pragma unroll
  for (int kt = 0; kt < KT1; ++kt) {
    const int kb = kt * 32;
    const float* rp = segp[kb >> 7] + (kb & 127) + khalf;
    bf16x16 a;
#pragma unroll
    for (int i = 0; i < 8; ++i) { a[i] = (__bf16)rp[i]; a[i + 8] = (__bf16)rp[i + 16]; }
#pragma unroll
    for (int nt = 0; nt < NT; ++nt) {
      const __bf16* bp = W1p + ((size_t)(kt * NT + nt) * 32 + lane) * 16;
      bf16x16 b;
#pragma unroll
      for (int i = 0; i < 16; ++i) b[i] = bp[i];
      acc[nt] = wmma_bf16(a, b, acc[nt]);
    }
  }

  // -------- bias + relu, stage through LDS as bf16 (transpose) --------
  __bf16* tbuf = lds_t[wv];
#pragma unroll
  for (int nt = 0; nt < NT; ++nt) {
    const float bias = b1[nt * 16 + ncol];
#pragma unroll
    for (int r = 0; r < 8; ++r) {
      float v = acc[nt][r] + bias;
      v = v > 0.f ? v : 0.f;
      tbuf[(mbase + r) * CCH + nt * 16 + ncol] = (__bf16)v;
    }
  }

  // -------- GEMM2: (16 x 128) @ (128 x 128), f32 accum --------
  f32x8 acc2[NT];
#pragma unroll
  for (int nt = 0; nt < NT; ++nt) acc2[nt] = (f32x8){0,0,0,0,0,0,0,0};

#pragma unroll
  for (int kt = 0; kt < 4; ++kt) {
    const __bf16* tp = tbuf + row * CCH + kt * 32 + khalf;
    bf16x16 a;
#pragma unroll
    for (int i = 0; i < 8; ++i) { a[i] = tp[i]; a[i + 8] = tp[i + 16]; }
#pragma unroll
    for (int nt = 0; nt < NT; ++nt) {
      const __bf16* bp = W2p + ((size_t)(kt * NT + nt) * 32 + lane) * 16;
      bf16x16 b;
#pragma unroll
      for (int i = 0; i < 16; ++i) b[i] = bp[i];
      acc2[nt] = wmma_bf16(a, b, acc2[nt]);
    }
  }

  // -------- epilogue: f32 residual + bias, in-place (own rows only) --------
#pragma unroll
  for (int nt = 0; nt < NT; ++nt) {
    const float bias = b2[nt * 16 + ncol];
#pragma unroll
    for (int r = 0; r < 8; ++r) {
      const int orow = tile * 16 + mbase + r;
      if (orow < nRows) {
        const size_t off = (size_t)orow * CCH + nt * 16 + ncol;
        io[off] = io[off] + acc2[nt][r] + bias;
      }
    }
  }
}

// ------------------------------ small helpers ------------------------------
__global__ void fill_f32(float* __restrict__ p, int n, float v) {
  int i = blockIdx.x * blockDim.x + threadIdx.x;
  if (i < n) p[i] = v;
}

__global__ void agg_edges(const float* __restrict__ e, const int* __restrict__ dst,
                          float* __restrict__ agg, int nE) {
  int i = blockIdx.x * blockDim.x + threadIdx.x;
  if (i >= nE * CCH) return;
  int ed = i >> 7, c = i & 127;
  atomicAdd(&agg[(size_t)dst[ed] * CCH + c], e[i]);
}

__global__ void pool_nodes(const float* __restrict__ x, const int* __restrict__ perm,
                           float* __restrict__ x1, int n) {
  int i = blockIdx.x * blockDim.x + threadIdx.x;
  if (i >= n * CCH) return;
  int nd = i >> 7, c = i & 127;
  atomicAdd(&x1[(size_t)perm[nd] * CCH + c], x[i]);
}

__global__ void scale_f32(float* __restrict__ p, int n, float s) {
  int i = blockIdx.x * blockDim.x + threadIdx.x;
  if (i < n) p[i] *= s;
}

__global__ void pool_edges(const float* __restrict__ e, const int* __restrict__ e2c,
                           float* __restrict__ ea1, int nE, int E1) {
  int i = blockIdx.x * blockDim.x + threadIdx.x;
  if (i >= nE * CCH) return;
  int ed = i >> 7, c = i & 127;
  int k = e2c[ed];
  if (k < E1) atomicAdd(&ea1[(size_t)k * CCH + c], e[i]);
}

__global__ void unpool_x(float* __restrict__ x, const float* __restrict__ x1,
                         const int* __restrict__ perm, int n) {
  int i = blockIdx.x * blockDim.x + threadIdx.x;
  if (i >= n * CCH) return;
  int nd = i >> 7, c = i & 127;
  x[i] += x1[(size_t)perm[nd] * CCH + c];
}

__global__ void unpool_e(float* __restrict__ e, const float* __restrict__ ea1,
                         const int* __restrict__ pe, int E1) {
  int i = blockIdx.x * blockDim.x + threadIdx.x;
  if (i >= E1 * CCH) return;
  int j = i >> 7, c = i & 127;
  e[(size_t)pe[j] * CCH + c] += ea1[i];
}

// ---------------------------------------------------------------------------
extern "C" void kernel_launch(void* const* d_in, const int* in_sizes, int n_in,
                              void* d_out, int out_size, void* d_ws, size_t ws_size,
                              hipStream_t stream) {
  const float* x_in  = (const float*)d_in[0];
  const float* ea_in = (const float*)d_in[1];
  const float* We1 = (const float*)d_in[2];
  const float* be1 = (const float*)d_in[3];
  const float* We2 = (const float*)d_in[4];
  const float* be2 = (const float*)d_in[5];
  const float* Wn1 = (const float*)d_in[6];
  const float* bn1 = (const float*)d_in[7];
  const float* Wn2 = (const float*)d_in[8];
  const float* bn2 = (const float*)d_in[9];
  const int* ei   = (const int*)d_in[10];
  const int* perm = (const int*)d_in[11];
  const int* pei  = (const int*)d_in[12];
  const int* e2c  = (const int*)d_in[13];
  const int* pe   = (const int*)d_in[14];

  const int N0 = in_sizes[0] / CCH;     // 16384
  const int E0 = in_sizes[10] / 2;      // 65024
  const int E1 = in_sizes[12] / 2;      // 3968
  const int NC = 1024;                  // (G/CL)^2 per reference
  const int L  = in_sizes[3] / CCH;     // 6

  const int* src0 = ei;       const int* dst0 = ei + E0;
  const int* srcP = pei;      const int* dstP = pei + E1;

  // ---- workspace carve (256B aligned) ----
  char* w = (char*)d_ws;
  auto carve = [&](size_t bytes) { void* p = (void*)w; w += (bytes + 255) & ~(size_t)255; return p; };
  float*  x_cur = (float*) carve((size_t)N0 * CCH * 4);
  float*  e_cur = (float*) carve((size_t)E0 * CCH * 4);
  float*  agg   = (float*) carve((size_t)N0 * CCH * 4);
  float*  x1    = (float*) carve((size_t)NC * CCH * 4);
  float*  ea1   = (float*) carve((size_t)E1 * CCH * 4);
  __bf16* W1p   = (__bf16*)carve((size_t)L * 3 * CCH * CCH * 2);
  __bf16* W2p   = (__bf16*)carve((size_t)L * CCH * CCH * 2);
  __bf16* Wn1p  = (__bf16*)carve((size_t)L * 2 * CCH * CCH * 2);
  __bf16* Wn2p  = (__bf16*)carve((size_t)L * CCH * CCH * 2);

  // working copies (inputs must stay pristine; graph replays are identical)
  hipMemcpyAsync(x_cur, x_in,  (size_t)N0 * CCH * 4, hipMemcpyDeviceToDevice, stream);
  hipMemcpyAsync(e_cur, ea_in, (size_t)E0 * CCH * 4, hipMemcpyDeviceToDevice, stream);

  // ---- pack weights to bf16 WMMA fragment layout ----
  for (int l = 0; l < L; ++l) {
    int n;
    n = 3 * CCH * CCH;
    pack_w<<<(n + 255) / 256, 256, 0, stream>>>(We1 + (size_t)l * n, W1p + (size_t)l * n, 3 * CCH, CCH);
    n = CCH * CCH;
    pack_w<<<(n + 255) / 256, 256, 0, stream>>>(We2 + (size_t)l * n, W2p + (size_t)l * n, CCH, CCH);
    n = 2 * CCH * CCH;
    pack_w<<<(n + 255) / 256, 256, 0, stream>>>(Wn1 + (size_t)l * n, Wn1p + (size_t)l * n, 2 * CCH, CCH);
    n = CCH * CCH;
    pack_w<<<(n + 255) / 256, 256, 0, stream>>>(Wn2 + (size_t)l * n, Wn2p + (size_t)l * n, CCH, CCH);
  }

  auto run_block = [&](float* xb, float* eb, const int* s, const int* d,
                       int nN, int nE, int l0, int l1) {
    for (int l = l0; l <= l1; ++l) {
      int eTiles = (nE + 15) / 16;
      fused_mlp<3><<<(eTiles + 3) / 4, 128, 0, stream>>>(
          eb, xb, s, d,
          W1p + (size_t)l * 3 * CCH * CCH, be1 + (size_t)l * CCH,
          W2p + (size_t)l * CCH * CCH,     be2 + (size_t)l * CCH, nE);
      int na = nN * CCH;
      fill_f32<<<(na + 255) / 256, 256, 0, stream>>>(agg, na, 0.f);
      int ne = nE * CCH;
      agg_edges<<<(ne + 255) / 256, 256, 0, stream>>>(eb, d, agg, nE);
      int nTiles = (nN + 15) / 16;
      fused_mlp<2><<<(nTiles + 3) / 4, 128, 0, stream>>>(
          xb, agg, nullptr, nullptr,
          Wn1p + (size_t)l * 2 * CCH * CCH, bn1 + (size_t)l * CCH,
          Wn2p + (size_t)l * CCH * CCH,     bn2 + (size_t)l * CCH, nN);
    }
  };

  // fine layers 0,1
  run_block(x_cur, e_cur, src0, dst0, N0, E0, 0, 1);

  // pool: x1 = segment_mean(x0, perm) (each cluster has exactly 16 members)
  {
    int n1 = NC * CCH;
    fill_f32<<<(n1 + 255) / 256, 256, 0, stream>>>(x1, n1, 0.f);
    int n0 = N0 * CCH;
    pool_nodes<<<(n0 + 255) / 256, 256, 0, stream>>>(x_cur, perm, x1, N0);
    scale_f32<<<(n1 + 255) / 256, 256, 0, stream>>>(x1, n1, 1.0f / 16.0f);
    int ne1 = E1 * CCH;
    fill_f32<<<(ne1 + 255) / 256, 256, 0, stream>>>(ea1, ne1, 0.f);
    int ne0 = E0 * CCH;
    pool_edges<<<(ne0 + 255) / 256, 256, 0, stream>>>(e_cur, e2c, ea1, E0, E1);
  }

  // coarse layers 2,3
  run_block(x1, ea1, srcP, dstP, NC, E1, 2, 3);

  // unpool
  {
    int n0 = N0 * CCH;
    unpool_x<<<(n0 + 255) / 256, 256, 0, stream>>>(x_cur, x1, perm, N0);
    int ne1 = E1 * CCH;
    unpool_e<<<(ne1 + 255) / 256, 256, 0, stream>>>(e_cur, ea1, pe, E1);
  }

  // fine layers 4,5
  run_block(x_cur, e_cur, src0, dst0, N0, E0, 4, 5);

  // outputs: (x, e) concatenated
  float* outx = (float*)d_out;
  hipMemcpyAsync(outx, x_cur, (size_t)N0 * CCH * 4, hipMemcpyDeviceToDevice, stream);
  hipMemcpyAsync(outx + (size_t)N0 * CCH, e_cur, (size_t)E0 * CCH * 4,
                 hipMemcpyDeviceToDevice, stream);
}
